// MoConoid_12618613916146
// MI455X (gfx1250) — matrix operand order repaired and checked by reading, not compiled, not run on hardware
//
#include <hip/hip_runtime.h>

typedef __attribute__((ext_vector_type(16))) _Float16 v16h;
typedef __attribute__((ext_vector_type(8)))  _Float16 v8h;
typedef __attribute__((ext_vector_type(4)))  _Float16 v4h;
typedef __attribute__((ext_vector_type(8)))  float    v8f;

#define D        768
#define NN       64
#define BS       512
#define THREADS  256

// ---- LDS layout (bytes), all 16B aligned ----
#define H_STRIDE    776   // halves, 64 rows
#define HPT_STRIDE  72    // halves, 768 rows
#define SLAB_STRIDE 40    // halves, 384 rows, x2 buffers
#define SLAB_HALVES (384 * SLAB_STRIDE)
#define ATTN_STRIDE 72    // halves, 64 rows
#define OFF_H     0
#define OFF_HPT   99328
#define OFF_SLAB  209920
#define OFF_ATTN  271360
#define OFF_SRC   280576
#define OFF_DST   280832
#define OFF_ROW0  281088
#define OFF_CC    284160
#define OFF_RED   287232
#define SMEM_BYTES 287296

static __device__ inline v8f v8f_splat(float x) {
  v8f v;
#pragma unroll
  for (int i = 0; i < 8; ++i) v[i] = x;
  return v;
}

// Async global->LDS 16B copy (GLOBAL_LOAD_ASYNC_TO_LDS_B128, ASYNCcnt tracked).
// LDS operand = wave-relative byte offset = low 32 bits of the generic pointer.
static __device__ inline void async_copy_b128(const void* gptr, const void* lptr) {
  unsigned loff = (unsigned)(unsigned long long)lptr;
  asm volatile("global_load_async_to_lds_b128 %0, %1, off"
               :: "v"(loff), "v"(gptr) : "memory");
}
static __device__ inline void wait_async0() {
  asm volatile("s_wait_asynccnt 0x0" ::: "memory");
}

// Convert W (f32, [f][o]) -> wt16 (f16, [o][f]) so WMMA B-fragments are contiguous.
__global__ void prep_wt(const float* __restrict__ w, _Float16* __restrict__ wt16) {
  int idx = blockIdx.x * 256 + threadIdx.x;
  if (idx < D * D) {
    int o = idx / D, f = idx % D;
    wt16[idx] = (_Float16)w[(size_t)f * D + o];
  }
}

// Convert mlp_w (f32, [j][k], 768x1536) -> f16 as-is (row j == output col, contiguous k).
__global__ void prep_wm(const float* __restrict__ mw, _Float16* __restrict__ wm16) {
  int idx = blockIdx.x * 256 + threadIdx.x;
  if (idx < D * 2 * D) wm16[idx] = (_Float16)mw[idx];
}

__global__ __launch_bounds__(THREADS, 1)
void gat_fused(const float* __restrict__ batch,
               const _Float16* __restrict__ wt16,
               const float* __restrict__ a_src,
               const float* __restrict__ a_dst,
               const float* __restrict__ gat_bias,
               const _Float16* __restrict__ wm16,
               const float* __restrict__ mlp_b,
               float* __restrict__ out)
{
  extern __shared__ char smem[];
  _Float16* shH    = (_Float16*)(smem + OFF_H);     // h, row-major [64][776]
  _Float16* shHPT  = (_Float16*)(smem + OFF_HPT);   // hp transposed [768][72]
  _Float16* shSlab = (_Float16*)(smem + OFF_SLAB);  // Wt K-slab, 2x [384][40]
  _Float16* shAttn = (_Float16*)(smem + OFF_ATTN);  // attn [64][72]
  float*    shSrc  = (float*)(smem + OFF_SRC);
  float*    shDst  = (float*)(smem + OFF_DST);
  float*    shRow0 = (float*)(smem + OFF_ROW0);     // f32 row0 of h / MLP out
  _Float16* shCC   = (_Float16*)(smem + OFF_CC);    // concat vec [1536]
  float*    shRed  = (float*)(smem + OFF_RED);

  const int tid  = threadIdx.x;
  const int wave = tid >> 5;
  const int lane = tid & 31;
  const int lr   = lane & 15;
  const int lhi  = lane >> 4;
  const int b    = blockIdx.x;
  const float* xb = batch + (size_t)b * NN * (D + NN);

  // ---- Phase 0: features (f32) -> shH (f16) ----
  for (int i = wave; i < NN; i += 8) {
    const float4* srow = (const float4*)(xb + (size_t)i * (D + NN));
    for (int c = lane; c < D / 4; c += 32) {
      float4 v = srow[c];
      v4h hv;
      hv[0] = (_Float16)v.x; hv[1] = (_Float16)v.y;
      hv[2] = (_Float16)v.z; hv[3] = (_Float16)v.w;
      *(v4h*)(shH + i * H_STRIDE + c * 4) = hv;
    }
  }
  __syncthreads();

  const int m = wave & 3;   // M-tile (node rows) for GEMM1
  const int g = wave >> 2;  // N-half within pass

  for (int layer = 0; layer < 2; ++layer) {
    // ---- GEMM1: hp = h(64x768) @ W(768x768), written transposed into shHPT ----
    for (int p = 0; p < 2; ++p) {
      v8f acc[12];
#pragma unroll
      for (int t = 0; t < 12; ++t) acc[t] = v8f_splat(0.f);

      // Prologue: async-stage slab for ks=0 into buffer 0.
      {
#pragma unroll
        for (int j = 0; j < 6; ++j) {
          int c = tid + j * THREADS;
          int row = c >> 2, kc = (c & 3) << 3;
          async_copy_b128(wt16 + (size_t)(p * 384 + row) * D + kc,
                          shSlab + row * SLAB_STRIDE + kc);
        }
      }

      for (int ks = 0; ks < 24; ++ks) {
        const int cur = ks & 1;
        const int k0  = ks * 32;
        wait_async0();        // our part of buf[cur] has landed
        __syncthreads();      // everyone's part has landed; buf[cur^1] readers done
        if (ks + 1 < 24) {    // async-stage next slab into the other buffer
          _Float16* nbuf = shSlab + (cur ^ 1) * SLAB_HALVES;
          const int nk0 = (ks + 1) * 32;
#pragma unroll
          for (int j = 0; j < 6; ++j) {
            int c = tid + j * THREADS;
            int row = c >> 2, kc = (c & 3) << 3;
            async_copy_b128(wt16 + (size_t)(p * 384 + row) * D + nk0 + kc,
                            nbuf + row * SLAB_STRIDE + kc);
          }
        }
        const _Float16* slab = shSlab + cur * SLAB_HALVES;

        // A fragment: rows m*16.., K = k0..k0+32 (per-lane two b128 chunks)
        v16h afr;
        const _Float16* arow = shH + (m * 16 + lr) * H_STRIDE + k0 + lhi * 8;
        ((v8h*)&afr)[0] = *(const v8h*)(arow);
        ((v8h*)&afr)[1] = *(const v8h*)(arow + 16);

#pragma unroll
        for (int t = 0; t < 12; ++t) {
          const _Float16* bp =
              slab + (g * 192 + t * 16 + lr) * SLAB_STRIDE + lhi * 8;
          v16h bfr;
          ((v8h*)&bfr)[0] = *(const v8h*)(bp);
          ((v8h*)&bfr)[1] = *(const v8h*)(bp + 16);
          acc[t] = __builtin_amdgcn_wmma_f32_16x16x32_f16(
              false, afr, false, bfr, (short)0, acc[t], false, false);
        }
      }
      // Epilogue: f32 tile -> f16 transposed hpT[o][i]; per-lane rows contiguous
#pragma unroll
      for (int t = 0; t < 12; ++t) {
        int o = (p * 24 + g * 12 + t) * 16 + lr;
        v8h hv;
#pragma unroll
        for (int r = 0; r < 8; ++r) hv[r] = (_Float16)acc[t][r];
        *(v8h*)(shHPT + o * HPT_STRIDE + m * 16 + lhi * 8) = hv;
      }
    }
    __syncthreads();

    // ---- tanh(hp) . a_src / a_dst reductions ----
    if (tid < NN) { shSrc[tid] = 0.f; shDst[tid] = 0.f; }
    __syncthreads();
    {
      int i = tid & 63;
      int q = tid >> 6;  // quarter of the 768 features
      float s = 0.f, d = 0.f;
      for (int o = q * 192; o < q * 192 + 192; ++o) {
        float v = tanhf((float)shHPT[o * HPT_STRIDE + i]);
        s += v * a_src[o];
        d += v * a_dst[o];
      }
      atomicAdd(&shSrc[i], s);
      atomicAdd(&shDst[i], d);
    }
    __syncthreads();

    // ---- attention: leaky-relu, mask, row softmax -> shAttn (f16) ----
    if (tid < NN) {
      int i = tid;
      const float* adjrow = xb + (size_t)i * (D + NN) + D;
      float si = shSrc[i];
      float mx = -3.0e38f;
      for (int j = 0; j < NN; ++j) {
        float a = si + shDst[j];
        a = a >= 0.f ? a : 0.2f * a;
        bool ok = (adjrow[j] != 0.f) || (j == i);
        if (ok && a > mx) mx = a;
      }
      float sum = 0.f;
      for (int j = 0; j < NN; ++j) {
        float a = si + shDst[j];
        a = a >= 0.f ? a : 0.2f * a;
        bool ok = (adjrow[j] != 0.f) || (j == i);
        float e = ok ? expf(a - mx) : 0.f;
        sum += e;
        shAttn[i * ATTN_STRIDE + j] = (_Float16)e;
      }
      float inv = 1.f / sum;
      for (int j = 0; j < NN; ++j)
        shAttn[i * ATTN_STRIDE + j] =
            (_Float16)((float)shAttn[i * ATTN_STRIDE + j] * inv);
    }
    __syncthreads();

    // ---- GEMM2: out = attn(64x64) @ hp(64x768) + bias -> shH (in place) ----
    for (int u = 0; u < 24; ++u) {
      int id = wave * 24 + u;
      int mm = id / 48;
      int n  = id % 48;
      float bv = gat_bias[n * 16 + lr];
      v8f acc = v8f_splat(bv);
#pragma unroll
      for (int k0 = 0; k0 < NN; k0 += 32) {
        v16h afr;
        const _Float16* arow = shAttn + (mm * 16 + lr) * ATTN_STRIDE + k0 + lhi * 8;
        ((v8h*)&afr)[0] = *(const v8h*)(arow);
        ((v8h*)&afr)[1] = *(const v8h*)(arow + 16);
        const _Float16* bp = shHPT + (n * 16 + lr) * HPT_STRIDE + k0 + lhi * 8;
        v16h bfr;
        ((v8h*)&bfr)[0] = *(const v8h*)(bp);
        ((v8h*)&bfr)[1] = *(const v8h*)(bp + 16);
        acc = __builtin_amdgcn_wmma_f32_16x16x32_f16(
            false, afr, false, bfr, (short)0, acc, false, false);
      }
#pragma unroll
      for (int r = 0; r < 8; ++r)
        shH[(mm * 16 + lhi * 8 + r) * H_STRIDE + n * 16 + lr] = (_Float16)acc[r];
      if (mm == 0 && lhi == 0)        // keep row 0 in f32 for l2norm
        shRow0[n * 16 + lr] = acc[0];
    }
    __syncthreads();
  }  // layer

  // ---- l2norm(h[:,0]) and concat vector ----
  if (tid == 0) shRed[0] = 0.f;
  __syncthreads();
  {
    float ss = 0.f;
    for (int o = tid; o < D; o += THREADS) { float v = shRow0[o]; ss += v * v; }
    atomicAdd(&shRed[0], ss);
  }
  __syncthreads();
  float inv1 = 1.f / fmaxf(sqrtf(shRed[0]), 1e-12f);
  for (int o = tid; o < D; o += THREADS) {
    shCC[o]     = (_Float16)xb[o];                 // center = batch[b][0][:D]
    shCC[D + o] = (_Float16)(shRow0[o] * inv1);    // normalized neighborhood
  }
  __syncthreads();

  // ---- MLP: (1x1536) @ mlp_w^T via WMMA (row 0 valid, others masked to 0) ----
  unsigned amask = (lr == 0) ? 0xFFFFFFFFu : 0u;
  v8f macc[6];
#pragma unroll
  for (int t = 0; t < 6; ++t) macc[t] = v8f_splat(0.f);
  for (int k0 = 0; k0 < 2 * D; k0 += 32) {
    v16h afr;
    ((v8h*)&afr)[0] = *(const v8h*)(shCC + k0 + lhi * 8);
    ((v8h*)&afr)[1] = *(const v8h*)(shCC + k0 + 16 + lhi * 8);
    unsigned* au = (unsigned*)&afr;
#pragma unroll
    for (int q = 0; q < 8; ++q) au[q] &= amask;   // zero A rows 1..15, EXEC stays full
#pragma unroll
    for (int t = 0; t < 6; ++t) {
      int n = wave * 6 + t;
      const _Float16* bp = wm16 + (size_t)(n * 16 + lr) * (2 * D) + k0 + lhi * 8;
      v16h bfr;
      ((v8h*)&bfr)[0] = *(const v8h*)(bp);
      ((v8h*)&bfr)[1] = *(const v8h*)(bp + 16);
      macc[t] = __builtin_amdgcn_wmma_f32_16x16x32_f16(
          false, afr, false, bfr, (short)0, macc[t], false, false);
    }
  }
  __syncthreads();
#pragma unroll
  for (int t = 0; t < 6; ++t) {
    int n = wave * 6 + t;
    if (lhi == 0)                                   // row 0 lives in VGPR0, lanes 0..15
      shRow0[n * 16 + lr] = macc[t][0] + mlp_b[n * 16 + lr];
  }
  __syncthreads();

  // ---- final l2norm + store ----
  if (tid == 0) shRed[0] = 0.f;
  __syncthreads();
  {
    float ss = 0.f;
    for (int o = tid; o < D; o += THREADS) { float v = shRow0[o]; ss += v * v; }
    atomicAdd(&shRed[0], ss);
  }
  __syncthreads();
  float inv2 = 1.f / fmaxf(sqrtf(shRed[0]), 1e-12f);
  for (int o = tid; o < D; o += THREADS)
    out[(size_t)b * D + o] = shRow0[o] * inv2;
}

extern "C" void kernel_launch(void* const* d_in, const int* in_sizes, int n_in,
                              void* d_out, int out_size, void* d_ws, size_t ws_size,
                              hipStream_t stream) {
  (void)in_sizes; (void)n_in; (void)out_size; (void)ws_size;
  const float* batch  = (const float*)d_in[0];
  const float* w      = (const float*)d_in[1];
  const float* a_src  = (const float*)d_in[2];
  const float* a_dst  = (const float*)d_in[3];
  const float* gatb   = (const float*)d_in[4];
  const float* mlp_w  = (const float*)d_in[5];
  const float* mlp_b  = (const float*)d_in[6];
  float* out = (float*)d_out;

  _Float16* wt16 = (_Float16*)d_ws;                 // 768*768 f16
  _Float16* wm16 = wt16 + (size_t)D * D;            // 768*1536 f16

  prep_wt<<<(D * D + 255) / 256, 256, 0, stream>>>(w, wt16);
  prep_wm<<<(D * 2 * D + 255) / 256, 256, 0, stream>>>(mlp_w, wm16);
  gat_fused<<<BS, THREADS, SMEM_BYTES, stream>>>(batch, wt16, a_src, a_dst,
                                                 gatb, wm16, mlp_b, out);
}
